// DenseTf_16415365005593
// MI455X (gfx1250) — matrix operand order repaired and checked by reading, not compiled, not run on hardware
//
#include <hip/hip_runtime.h>

typedef __attribute__((ext_vector_type(16))) __bf16 v16bf;
typedef __attribute__((ext_vector_type(8)))  __bf16 v8bf;
typedef __attribute__((ext_vector_type(8)))  float  v8f;
typedef __attribute__((ext_vector_type(4)))  int    v4i;

#define M_DIM 8192   // tokens
#define N_DIM 4096   // filters
#define K_DIM 4096   // d_in

#define BM 128
#define BN 256
#define BK 32
#define LDS_PITCH 40   // 32 halves + 8 pad

#if defined(__has_builtin)
#  if __has_builtin(__builtin_amdgcn_global_load_async_to_lds_b128) && \
      __has_builtin(__builtin_amdgcn_s_wait_asynccnt)
#    define USE_ASYNC_LDS 1
#  endif
#endif
#ifndef USE_ASYNC_LDS
#  define USE_ASYNC_LDS 0
#endif

__device__ __forceinline__ unsigned short f32_to_bf16_rne(float f) {
  unsigned int u = __float_as_uint(f);
  u += 0x7FFFu + ((u >> 16) & 1u);   // round to nearest even
  return (unsigned short)(u >> 16);
}

// ---------------- prepass 1: x (f32) -> bf16, 8 elems/thread ----------------
__global__ void cvt_x_bf16(const float* __restrict__ x, unsigned short* __restrict__ xb) {
  size_t i = ((size_t)blockIdx.x * blockDim.x + threadIdx.x) * 8;
  float4 f0 = *(const float4*)(x + i);
  float4 f1 = *(const float4*)(x + i + 4);
  uint4 o;
  o.x = (unsigned)f32_to_bf16_rne(f0.x) | ((unsigned)f32_to_bf16_rne(f0.y) << 16);
  o.y = (unsigned)f32_to_bf16_rne(f0.z) | ((unsigned)f32_to_bf16_rne(f0.w) << 16);
  o.z = (unsigned)f32_to_bf16_rne(f1.x) | ((unsigned)f32_to_bf16_rne(f1.y) << 16);
  o.w = (unsigned)f32_to_bf16_rne(f1.z) | ((unsigned)f32_to_bf16_rne(f1.w) << 16);
  *(uint4*)(xb + i) = o;
}

// ---- prepass 2: w (f32 [K][N]) -> sign(w) as bf16, transposed to [N][K] ----
__global__ void binarize_transpose_w(const float* __restrict__ w, unsigned short* __restrict__ wt) {
  __shared__ unsigned short tile[32][33];
  const int f  = blockIdx.x * 32 + threadIdx.x;   // filter index (coalesced read)
  const int k0 = blockIdx.y * 32;
  #pragma unroll
  for (int rr = 0; rr < 4; ++rr) {
    int r = threadIdx.y + rr * 8;
    float v = w[(size_t)(k0 + r) * N_DIM + f];
    tile[r][threadIdx.x] = (v > 0.f) ? 0x3F80u : ((v < 0.f) ? 0xBF80u : 0u);
  }
  __syncthreads();
  const int k  = k0 + threadIdx.x;                // k index (coalesced write)
  const int f0 = blockIdx.x * 32;
  #pragma unroll
  for (int rr = 0; rr < 4; ++rr) {
    int r = threadIdx.y + rr * 8;
    wt[(size_t)(f0 + r) * K_DIM + k] = tile[threadIdx.x][r];
  }
}

// ---------------------------- WMMA GEMM ----------------------------
union FragBF {
  v16bf v;
  struct { v8bf lo, hi; } h;
};

#if USE_ASYNC_LDS
// Builtin signature (from clang diagnostic): (int4 AS1*, int4 AS3*, imm int, imm int)
typedef __attribute__((address_space(1))) v4i as1_v4i;
typedef __attribute__((address_space(3))) v4i as3_v4i;
#define GLD_ASYNC_B128(gp, lp)                                        \
  __builtin_amdgcn_global_load_async_to_lds_b128(                     \
      (as1_v4i*)(gp), (as3_v4i*)(lp), 0, 0)
#endif

__launch_bounds__(256)
__attribute__((amdgpu_waves_per_eu(4)))   // cap VGPRs to low bank, 4 waves/SIMD
__global__ void gemm_bin_wmma(const unsigned short* __restrict__ xb,   // [M][K] bf16
                              const unsigned short* __restrict__ wt,   // [N][K] bf16 (+/-1)
                              const float* __restrict__ bias,
                              float* __restrict__ out) {               // [M][N] f32
  __shared__ unsigned short As[2][BM][LDS_PITCH];   // 20 KiB
  __shared__ unsigned short Bs[2][BN][LDS_PITCH];   // 40 KiB

  const int tid   = threadIdx.x;
  const int lane  = tid & 31;
  const int wave  = tid >> 5;        // 0..7
  const int wm    = wave >> 2;       // 0..1 -> 64-row M band
  const int wn    = wave & 3;        // 0..3 -> 64-col N band
  const int lrow  = lane & 15;
  const int lhalf = lane >> 4;

  const int m_block = blockIdx.y * BM;
  const int n_block = blockIdx.x * BN;

  // staging: each thread owns (rbase + {0,64}) rows of A and (rbase + {0,64,128,192})
  // rows of B, one 16B chunk at column koff.
  const int koff  = (tid & 3) * 8;
  const int rbase = tid >> 2;        // 0..63
  const unsigned short* gA = xb + (size_t)(m_block + rbase) * K_DIM + koff;
  const unsigned short* gB = wt + (size_t)(n_block + rbase) * K_DIM + koff;

  v8f acc[4][4];
  #pragma unroll
  for (int i = 0; i < 4; ++i)
    #pragma unroll
    for (int j = 0; j < 4; ++j)
      acc[i][j] = (v8f){0.f, 0.f, 0.f, 0.f, 0.f, 0.f, 0.f, 0.f};

  const int KSTEPS = K_DIM / BK;

#if USE_ASYNC_LDS
  // -------- CDNA5 async global->LDS staging (no staging VGPRs, ASYNCcnt) -----
  auto stage_async = [&](int buf, int kk) {
    const size_t kb = (size_t)kk * BK;
    GLD_ASYNC_B128(gA + kb,                         &As[buf][rbase      ][koff]);
    GLD_ASYNC_B128(gA + kb + (size_t)64  * K_DIM,   &As[buf][rbase + 64 ][koff]);
    GLD_ASYNC_B128(gB + kb,                         &Bs[buf][rbase      ][koff]);
    GLD_ASYNC_B128(gB + kb + (size_t)64  * K_DIM,   &Bs[buf][rbase + 64 ][koff]);
    GLD_ASYNC_B128(gB + kb + (size_t)128 * K_DIM,   &Bs[buf][rbase + 128][koff]);
    GLD_ASYNC_B128(gB + kb + (size_t)192 * K_DIM,   &Bs[buf][rbase + 192][koff]);
  };

  stage_async(0, 0);
  __builtin_amdgcn_s_wait_asynccnt(0);
  __syncthreads();
#else
  // ----------------- fallback: register staging via ds_store -----------------
  uint4 st[6];
  auto load_slab = [&](int kk) {
    const size_t kb = (size_t)kk * BK;
    st[0] = *(const uint4*)(gA + kb);
    st[1] = *(const uint4*)(gA + kb + (size_t)64  * K_DIM);
    st[2] = *(const uint4*)(gB + kb);
    st[3] = *(const uint4*)(gB + kb + (size_t)64  * K_DIM);
    st[4] = *(const uint4*)(gB + kb + (size_t)128 * K_DIM);
    st[5] = *(const uint4*)(gB + kb + (size_t)192 * K_DIM);
  };
  auto store_slab = [&](int buf) {
    *(uint4*)&As[buf][rbase      ][koff] = st[0];
    *(uint4*)&As[buf][rbase + 64 ][koff] = st[1];
    *(uint4*)&Bs[buf][rbase      ][koff] = st[2];
    *(uint4*)&Bs[buf][rbase + 64 ][koff] = st[3];
    *(uint4*)&Bs[buf][rbase + 128][koff] = st[4];
    *(uint4*)&Bs[buf][rbase + 192][koff] = st[5];
  };

  load_slab(0);
  store_slab(0);
  __syncthreads();
#endif

  // No unrolling: one loop body keeps one set of fragment registers live and
  // lets the accumulators keep a single stable (in-place) assignment.
  #pragma clang loop unroll(disable)
  for (int kk = 0; kk < KSTEPS; ++kk) {
    const int buf = kk & 1;

#if USE_ASYNC_LDS
    if (kk + 1 < KSTEPS) stage_async(buf ^ 1, kk + 1);
#else
    if (kk + 1 < KSTEPS) load_slab(kk + 1);
#endif

    // A fragments stay resident (32 VGPRs)
    FragBF af[4];
    #pragma unroll
    for (int i = 0; i < 4; ++i) {
      const unsigned short* p = &As[buf][wm * 64 + i * 16 + lrow][lhalf * 8];
      af[i].h.lo = *(const v8bf*)p;          // K = lhalf?8..15:0..7
      af[i].h.hi = *(const v8bf*)(p + 16);   // K = +16
    }

    // B fragments: 2-deep pipeline so the next LDS load overlaps the current
    // WMMA quartet (breaks the WAR chain on a single B register set).
    FragBF bfr[2];
    {
      const unsigned short* q0 = &Bs[buf][wn * 64 + lrow][lhalf * 8];
      bfr[0].h.lo = *(const v8bf*)q0;        // K = lhalf?8..15:0..7
      bfr[0].h.hi = *(const v8bf*)(q0 + 16); // K = +16
    }
    #pragma unroll
    for (int j = 0; j < 4; ++j) {
      if (j < 3) {
        const unsigned short* qn = &Bs[buf][wn * 64 + (j + 1) * 16 + lrow][lhalf * 8];
        bfr[(j + 1) & 1].h.lo = *(const v8bf*)qn;
        bfr[(j + 1) & 1].h.hi = *(const v8bf*)(qn + 16);
      }
      #pragma unroll
      for (int i = 0; i < 4; ++i)
        acc[i][j] = __builtin_amdgcn_wmma_f32_16x16x32_bf16(
            false, af[i].v, false, bfr[j & 1].v, (short)0, acc[i][j], false, false);
    }

    if (kk + 1 < KSTEPS) {
#if USE_ASYNC_LDS
      __builtin_amdgcn_s_wait_asynccnt(0);
      __syncthreads();
#else
      __syncthreads();
      store_slab(buf ^ 1);
      __syncthreads();
#endif
    }
  }

  // epilogue: C/D layout -> VGPR e holds row m0 + lhalf*8 + e, col n0 + lrow
  #pragma unroll
  for (int i = 0; i < 4; ++i) {
    #pragma unroll
    for (int j = 0; j < 4; ++j) {
      const int gm = m_block + wm * 64 + i * 16 + lhalf * 8;
      const int gn = n_block + wn * 64 + j * 16 + lrow;
      const float bb = bias[gn];
      #pragma unroll
      for (int e = 0; e < 8; ++e)
        out[(size_t)(gm + e) * N_DIM + gn] = acc[i][j][e] + bb;
    }
  }
}

extern "C" void kernel_launch(void* const* d_in, const int* in_sizes, int n_in,
                              void* d_out, int out_size, void* d_ws, size_t ws_size,
                              hipStream_t stream) {
  (void)in_sizes; (void)n_in; (void)out_size; (void)ws_size;
  const float* x = (const float*)d_in[0];
  const float* w = (const float*)d_in[1];
  const float* b = (const float*)d_in[2];
  float* out = (float*)d_out;

  unsigned short* xb = (unsigned short*)d_ws;                 // 64 MiB bf16 x
  unsigned short* wt = xb + (size_t)M_DIM * K_DIM;            // 32 MiB bf16 sign(w)^T

  const size_t xElems = (size_t)M_DIM * K_DIM;
  cvt_x_bf16<<<(unsigned)(xElems / 8 / 256), 256, 0, stream>>>(x, xb);
  binarize_transpose_w<<<dim3(N_DIM / 32, K_DIM / 32), dim3(32, 8), 0, stream>>>(w, wt);
  gemm_bin_wmma<<<dim3(N_DIM / BN, M_DIM / BM), 256, 0, stream>>>(xb, wt, b, out);
}
